// SaccadicLayer_4423816315043
// MI455X (gfx1250) — compile-verified, matrix operand order checked by reading, not run on hardware
//
#include <hip/hip_runtime.h>
#include <hip/hip_bf16.h>
#include <math.h>

// ---------------------------------------------------------------------------
// SaccadicLayer forward for MI455X (gfx1250, wave32, WMMA).
// All large matmuls -> v_wmma_f32_16x16x32_f16 (f16 in, f32 accum) with
// LDS-staged, double-buffered tiles (async global->LDS when available).
// Small sequential control path -> wave32 VALU kernels with shfl reductions.
// Workspace requirement: ~380 MB.
// ---------------------------------------------------------------------------

#define D_      512
#define NH_     8
#define HD_     64
#define NS_     4
#define BSZ     8        // peripheral token-block size
#define WS_     64       // saccade window
#define BB      8        // batch
#define NN      4096     // sequence length
#define NB_     512      // NN / BSZ
#define CTXMAX  320      // padded ctx rows per batch (1 + 4*64 = 257 used)
#define FF_     2048     // 4*D
#define KTILE   32       // K per WMMA step
#define KPAD    40       // padded halves per LDS row (80B stride = 20 banks,
                         // 20*r mod 64 distinct for r=0..15 -> conflict-free)

#ifndef __has_builtin
#define __has_builtin(x) 0
#endif
#if __has_builtin(__builtin_amdgcn_global_load_async_to_lds_b128) && \
    __has_builtin(__builtin_amdgcn_s_wait_asynccnt)
#define ATH_ASYNC 1
#else
#define ATH_ASYNC 0
#endif

typedef __attribute__((ext_vector_type(16))) _Float16 v16h;
typedef __attribute__((ext_vector_type(8)))  _Float16 v8h;
typedef __attribute__((ext_vector_type(8)))  float    v8f;
typedef __attribute__((ext_vector_type(4)))  int      v4i;

__device__ __forceinline__ float gelu_exact(float x) {
  return 0.5f * x * (1.0f + erff(x * 0.70710678118654752f));
}

__device__ __forceinline__ float wave_sum(float v) {
#pragma unroll
  for (int o = 16; o > 0; o >>= 1) v += __shfl_xor(v, o, 32);
  return v;
}

#if ATH_ASYNC
// gfx1250 async DMA: LDS[l] = MEM[g] (16B per lane), tracked by ASYNCcnt.
// Builtin prototype (from compiler diagnostic): (int4 AS1*, int4 AS3*, Ii, Ii).
__device__ __forceinline__ void async_cp_b128(const _Float16* g, _Float16* l) {
  __builtin_amdgcn_global_load_async_to_lds_b128(
      (__attribute__((address_space(1))) v4i*)(g),
      (__attribute__((address_space(3))) v4i*)(l), 0, 0);
}
#endif

// ---------------------------------------------------------------------------
// WMMA 16x32 f16 fragment from an LDS tile (rows padded to KPAD halves).
// ISA 7.12.2 layout: lane&15 = matrix row, lane>>4 selects K halves:
// elements 0..7 -> K = half*8..+7 ; elements 8..15 -> K = 16+half*8..+7.
// Same packing serves B (32x16) fragments: column n of B == row n of W.
// ---------------------------------------------------------------------------
__device__ __forceinline__ v16h lds_frag(const _Float16* base, int row, int lane) {
  const int m = lane & 15, half = lane >> 4;
  const _Float16* p = base + (row + m) * KPAD + half * 8;
  v8h lo = *(const v8h*)p;
  v8h hi = *(const v8h*)(p + 16);
  return __builtin_shufflevector(lo, hi, 0,1,2,3,4,5,6,7,8,9,10,11,12,13,14,15);
}

// ---------------------------------------------------------------------------
// Generic WMMA GEMM: C[M,N] = act(A[M,K] @ W[N,K]^T + bias) (+resid)
// Block = 256 threads (8 waves). Tile 128M x 64N x 32K, double-buffered LDS
// staging so the next tile's global->LDS traffic overlaps 32 WMMAs.
// Grid = (N/64, M/128). M mult of 128, N of 64, K of 32 (buffers padded).
// ---------------------------------------------------------------------------
__global__ __launch_bounds__(256)
void gemm_wmma_kernel(const _Float16* __restrict__ A, int lda,
                      const _Float16* __restrict__ W, int ldw,
                      const float* __restrict__ bias,
                      const float* __restrict__ resid, long ldres,
                      float* __restrict__ Cf, _Float16* __restrict__ Ch,
                      int ldc, int K, int act) {
  __shared__ alignas(16) _Float16 lA[2][128 * KPAD];
  __shared__ alignas(16) _Float16 lW[2][64 * KPAD];
  const int tid = threadIdx.x;
  const int lane = tid & 31;
  const int wave = tid >> 5;
  const int rowBase = blockIdx.y * 128;
  const int colBase = blockIdx.x * 64;

  // Stage one 128x32 A tile (512 x 16B chunks: 2/thread) and one 64x32 W tile
  // (256 x 16B chunks: 1/thread) into LDS buffer `buf`.
  auto stage = [&](int buf, int k0) {
#pragma unroll
    for (int q = 0; q < 2; ++q) {
      const int c = tid * 2 + q, r = c >> 2, p = c & 3;
      const _Float16* g = A + (size_t)(rowBase + r) * lda + k0 + p * 8;
      _Float16* l = &lA[buf][r * KPAD + p * 8];
#if ATH_ASYNC
      async_cp_b128(g, l);
#else
      *(v8h*)l = *(const v8h*)g;
#endif
    }
    {
      const int r = tid >> 2, p = tid & 3;
      const _Float16* g = W + (size_t)(colBase + r) * ldw + k0 + p * 8;
      _Float16* l = &lW[buf][r * KPAD + p * 8];
#if ATH_ASYNC
      async_cp_b128(g, l);
#else
      *(v8h*)l = *(const v8h*)g;
#endif
    }
  };

  v8f acc0 = {}, acc1 = {}, acc2 = {}, acc3 = {};
  const int nk = K / KTILE;
  stage(0, 0);
#if ATH_ASYNC
  __builtin_amdgcn_s_wait_asynccnt(0);
#endif
  __syncthreads();
  for (int ki = 0; ki < nk; ++ki) {
    const int cur = ki & 1;
    if (ki + 1 < nk) stage(cur ^ 1, (ki + 1) * KTILE);  // overlap with WMMAs
    v16h a  = lds_frag(&lA[cur][0], wave * 16, lane);
    v16h b0 = lds_frag(&lW[cur][0],  0, lane);
    v16h b1 = lds_frag(&lW[cur][0], 16, lane);
    v16h b2 = lds_frag(&lW[cur][0], 32, lane);
    v16h b3 = lds_frag(&lW[cur][0], 48, lane);
    acc0 = __builtin_amdgcn_wmma_f32_16x16x32_f16(false, a, false, b0, (short)0, acc0, false, false);
    acc1 = __builtin_amdgcn_wmma_f32_16x16x32_f16(false, a, false, b1, (short)0, acc1, false, false);
    acc2 = __builtin_amdgcn_wmma_f32_16x16x32_f16(false, a, false, b2, (short)0, acc2, false, false);
    acc3 = __builtin_amdgcn_wmma_f32_16x16x32_f16(false, a, false, b3, (short)0, acc3, false, false);
#if ATH_ASYNC
    __builtin_amdgcn_s_wait_asynccnt(0);  // my next-tile DMA complete
#endif
    __syncthreads();                      // publish across waves; WAR on cur
  }
  // C layout: lanes 0-15 -> N=lane, M=vgpr; lanes 16-31 -> N=lane-16, M=vgpr+8
  const int mrow = rowBase + wave * 16 + (lane >> 4) * 8;
  const int c0 = colBase + (lane & 15);
  v8f accs[4] = {acc0, acc1, acc2, acc3};
#pragma unroll
  for (int nt = 0; nt < 4; ++nt) {
    const int c = c0 + nt * 16;
    const float bb = bias ? bias[c] : 0.f;
#pragma unroll
    for (int r = 0; r < 8; ++r) {
      const int m = mrow + r;
      float v = accs[nt][r] + bb;
      if (act) v = gelu_exact(v);
      if (resid) v += resid[(size_t)m * ldres + c];
      const size_t o = (size_t)m * ldc + c;
      if (Cf) Cf[o] = v;
      if (Ch) Ch[o] = (_Float16)v;
    }
  }
}

// ---------------------------------------------------------------------------
// Row LayerNorm over D=512, optional +pos[row % posmod], f32/f16 outputs.
// 128 threads per row (4 waves x 4 elems/thread).
// ---------------------------------------------------------------------------
__global__ __launch_bounds__(128)
void ln512_kernel(const float* __restrict__ X, const float* __restrict__ g,
                  const float* __restrict__ bta, float* __restrict__ Yf,
                  _Float16* __restrict__ Yh, const float* __restrict__ pos,
                  int posmod) {
  __shared__ float red[8];
  const int row = blockIdx.x;
  const float* x = X + (size_t)row * D_;
  float v[4]; float s = 0.f;
#pragma unroll
  for (int i = 0; i < 4; ++i) { v[i] = x[threadIdx.x + i * 128]; s += v[i]; }
  s = wave_sum(s);
  if ((threadIdx.x & 31) == 0) red[threadIdx.x >> 5] = s;
  __syncthreads();
  const float mean = (red[0] + red[1] + red[2] + red[3]) * (1.f / D_);
  float s2 = 0.f;
#pragma unroll
  for (int i = 0; i < 4; ++i) { const float d = v[i] - mean; s2 += d * d; }
  s2 = wave_sum(s2);
  if ((threadIdx.x & 31) == 0) red[4 + (threadIdx.x >> 5)] = s2;
  __syncthreads();
  const float var = (red[4] + red[5] + red[6] + red[7]) * (1.f / D_);
  const float rs = rsqrtf(var + 1e-5f);
#pragma unroll
  for (int i = 0; i < 4; ++i) {
    const int c = threadIdx.x + i * 128;
    float y = (v[i] - mean) * rs * g[c] + bta[c];
    if (pos) y += pos[(size_t)(row % posmod) * D_ + c];
    const size_t o = (size_t)row * D_ + c;
    if (Yf) Yf[o] = y;
    if (Yh) Yh[o] = (_Float16)y;
  }
}

// ---------------------------------------------------------------------------
// Peripheral block stats: per (b, block of 8 tokens) -> softmax token weights,
// weighted mean/std + max over the block, write [mn|sd|mx] f16 row (1536).
// ---------------------------------------------------------------------------
__global__ __launch_bounds__(64)
void peripheral_kernel(const float* __restrict__ h, const float* __restrict__ w_w,
                       const float* __restrict__ w_b, _Float16* __restrict__ featin) {
  __shared__ float wt[BSZ];
  __shared__ float red2[2];
  const int blk = blockIdx.x;  // b*NB + j
  const float* base = h + (size_t)blk * BSZ * D_;
  for (int r = 0; r < BSZ; ++r) {
    float p = 0.f;
    for (int i = threadIdx.x; i < D_; i += 64) p += base[r * D_ + i] * w_w[i];
    p = wave_sum(p);
    if ((threadIdx.x & 31) == 0) red2[threadIdx.x >> 5] = p;
    __syncthreads();
    if (threadIdx.x == 0) wt[r] = red2[0] + red2[1] + w_b[0];
    __syncthreads();
  }
  if (threadIdx.x == 0) {
    float mx = wt[0];
    for (int r = 1; r < BSZ; ++r) mx = fmaxf(mx, wt[r]);
    float sum = 0.f;
    for (int r = 0; r < BSZ; ++r) { wt[r] = __expf(wt[r] - mx); sum += wt[r]; }
    const float inv = 1.f / sum;
    for (int r = 0; r < BSZ; ++r) wt[r] *= inv;
  }
  __syncthreads();
  _Float16* outr = featin + (size_t)blk * (3 * D_);
  for (int d = threadIdx.x; d < D_; d += 64) {
    float mn = 0.f, mx = -3.4e38f;
    for (int r = 0; r < BSZ; ++r) {
      const float xv = base[r * D_ + d];
      mn += wt[r] * xv; mx = fmaxf(mx, xv);
    }
    float var = 0.f;
    for (int r = 0; r < BSZ; ++r) {
      const float dd = base[r * D_ + d] - mn;
      var += wt[r] * dd * dd;
    }
    outr[d]          = (_Float16)mn;
    outr[D_ + d]     = (_Float16)sqrtf(var + 1e-8f);
    outr[2 * D_ + d] = (_Float16)mx;
  }
}

__global__ void st_mean_kernel(const float* __restrict__ pm, float* __restrict__ st) {
  const int idx = blockIdx.x * 256 + threadIdx.x;
  if (idx >= BB * D_) return;
  const int b = idx / D_, d = idx % D_;
  float s = 0.f;
  for (int j = 0; j < NB_; ++j) s += pm[((size_t)b * NB_ + j) * D_ + d];
  st[idx] = s * (1.f / NB_);
}

// Small f32 linear: Y[r,n] = act(X[r,:] . W[n,:] + bias[n]) (+resid[r*ldres+n])
__global__ void small_linear_kernel(const float* __restrict__ X, const float* __restrict__ W,
                                    const float* __restrict__ bias,
                                    const float* __restrict__ resid, long ldres,
                                    float* __restrict__ Y, int Mr, int Nout, int K, int act) {
  const int idx = blockIdx.x * 256 + threadIdx.x;
  if (idx >= Mr * Nout) return;
  const int rM = idx / Nout, n = idx % Nout;
  const float* x = X + (size_t)rM * K;
  const float* w = W + (size_t)n * K;
  float s = bias ? bias[n] : 0.f;
  for (int k = 0; k < K; ++k) s += x[k] * w[k];
  if (act) s = gelu_exact(s);
  if (resid) s += resid[(size_t)rM * ldres + n];
  Y[idx] = s;
}

// Controller: argmax_m dot(qs[b], kp[b,m]) (first index wins ties) -> start.
__global__ __launch_bounds__(256)
void saccade_select_kernel(const float* __restrict__ qs, const float* __restrict__ kp,
                           int* __restrict__ starts) {
  __shared__ float bv[256]; __shared__ int bix[256];
  const int b = blockIdx.x;
  const float* q = qs + (size_t)b * D_;
  float best = -3.4e38f; int bi = 1 << 30;
  for (int m = threadIdx.x; m < NB_; m += 256) {
    const float* kk = kp + ((size_t)b * NB_ + m) * D_;
    float s = 0.f;
    for (int d = 0; d < D_; ++d) s += q[d] * kk[d];
    if (s > best || (s == best && m < bi)) { best = s; bi = m; }
  }
  bv[threadIdx.x] = best; bix[threadIdx.x] = bi; __syncthreads();
  for (int off = 128; off > 0; off >>= 1) {
    if (threadIdx.x < off) {
      if (bv[threadIdx.x + off] > bv[threadIdx.x] ||
          (bv[threadIdx.x + off] == bv[threadIdx.x] &&
           bix[threadIdx.x + off] < bix[threadIdx.x])) {
        bv[threadIdx.x] = bv[threadIdx.x + off];
        bix[threadIdx.x] = bix[threadIdx.x + off];
      }
    }
    __syncthreads();
  }
  if (threadIdx.x == 0) {
    const int fp = bix[0] * BSZ;
    int s0 = fp - WS_ / 2;
    if (s0 < 0) s0 = 0;
    if (s0 > NN - WS_) s0 = NN - WS_;
    starts[b] = s0;
  }
}

__global__ void gather_win_kernel(const float* __restrict__ h, const int* __restrict__ starts,
                                  float* __restrict__ ctxf, _Float16* __restrict__ ctxh, int t) {
  const int idx = blockIdx.x * 256 + threadIdx.x;
  if (idx >= BB * WS_ * D_) return;
  const int b = idx / (WS_ * D_);
  const int i = (idx / D_) % WS_;
  const int d = idx % D_;
  const float v = h[((size_t)b * NN + starts[b] + i) * D_ + d];
  const size_t o = ((size_t)b * CTXMAX + 1 + t * WS_ + i) * D_ + d;
  ctxf[o] = v; ctxh[o] = (_Float16)v;
}

__global__ void set_ctx_st_kernel(const float* __restrict__ st,
                                  float* __restrict__ ctxf, _Float16* __restrict__ ctxh) {
  const int idx = blockIdx.x * 256 + threadIdx.x;
  if (idx >= BB * D_) return;
  const int b = idx / D_, d = idx % D_;
  const size_t o = ((size_t)b * CTXMAX) * D_ + d;
  const float v = st[idx];
  ctxf[o] = v; ctxh[o] = (_Float16)v;
}

// fov attention: only query row 0 of ctx matters (cl = ctx[:,0] afterwards).
// One wave per (b, head); online softmax over Lctx keys; qkv = [Q|K|V] rows.
__global__ __launch_bounds__(32)
void fov_attn_q0_kernel(const float* __restrict__ qkv, int Lctx, float* __restrict__ o0) {
  const int b = blockIdx.x >> 3, hh = blockIdx.x & 7;
  const int lane = threadIdx.x;
  const float* Q = qkv + (size_t)b * CTXMAX * (3 * D_) + hh * HD_;
  const float q0 = Q[lane * 2], q1 = Q[lane * 2 + 1];
  float mmax = -3.4e38f, denom = 0.f, a0 = 0.f, a1 = 0.f;
  for (int j = 0; j < Lctx; ++j) {
    const float* row = qkv + ((size_t)b * CTXMAX + j) * (3 * D_);
    const float* Kp = row + D_ + hh * HD_;
    float p = q0 * Kp[lane * 2] + q1 * Kp[lane * 2 + 1];
    p = wave_sum(p) * 0.125f;  // 1/sqrt(64)
    const float nm = fmaxf(mmax, p);
    const float corr = __expf(mmax - nm);
    const float e = __expf(p - nm);
    const float* Vp = row + 2 * D_ + hh * HD_;
    denom = denom * corr + e;
    a0 = a0 * corr + e * Vp[lane * 2];
    a1 = a1 * corr + e * Vp[lane * 2 + 1];
    mmax = nm;
  }
  const float inv = 1.f / denom;
  o0[(size_t)b * D_ + hh * HD_ + lane * 2]     = a0 * inv;
  o0[(size_t)b * D_ + hh * HD_ + lane * 2 + 1] = a1 * inv;
}

// Memory cross-attention: one wave per (b, m, head), keys = ac rows (ctx 1..Lk).
__global__ __launch_bounds__(256)
void ma_attn_kernel(const float* __restrict__ qma, const float* __restrict__ kma,
                    const float* __restrict__ vma, _Float16* __restrict__ ao, int Lk) {
  const int gw = blockIdx.x * 8 + (threadIdx.x >> 5);
  const int lane = threadIdx.x & 31;
  const int b = gw / (NB_ * NH_);
  const int m = (gw / NH_) % NB_;
  const int hh = gw % NH_;
  const float* q = qma + ((size_t)b * NB_ + m) * D_ + hh * HD_;
  const float q0 = q[lane * 2], q1 = q[lane * 2 + 1];
  float mmax = -3.4e38f, denom = 0.f, a0 = 0.f, a1 = 0.f;
  for (int j = 0; j < Lk; ++j) {
    const size_t r = ((size_t)b * CTXMAX + 1 + j) * D_ + hh * HD_;
    float p = q0 * kma[r + lane * 2] + q1 * kma[r + lane * 2 + 1];
    p = wave_sum(p) * 0.125f;
    const float nm = fmaxf(mmax, p);
    const float corr = __expf(mmax - nm);
    const float e = __expf(p - nm);
    denom = denom * corr + e;
    a0 = a0 * corr + e * vma[r + lane * 2];
    a1 = a1 * corr + e * vma[r + lane * 2 + 1];
    mmax = nm;
  }
  const float inv = 1.f / denom;
  const size_t o = ((size_t)b * NB_ + m) * D_ + hh * HD_;
  ao[o + lane * 2]     = (_Float16)(a0 * inv);
  ao[o + lane * 2 + 1] = (_Float16)(a1 * inv);
}

__global__ __launch_bounds__(32)
void gate_kernel(const float* __restrict__ mg1w, const float* __restrict__ mg1b,
                 const float* __restrict__ mg2w, const float* __restrict__ mg2b,
                 float tval, float* __restrict__ gate) {
  float s = 0.f;
  for (int d = threadIdx.x; d < D_; d += 32)
    s += gelu_exact(tval * mg1w[d] + mg1b[d]) * mg2w[d];
  s = wave_sum(s) + mg2b[0];
  if (threadIdx.x == 0) gate[0] = 1.f / (1.f + __expf(-s));
}

__global__ void pm_update_kernel(float* __restrict__ pm, _Float16* __restrict__ pmh,
                                 const float* __restrict__ dctx, const float* __restrict__ gate) {
  const int idx = blockIdx.x * 256 + threadIdx.x;
  if (idx >= BB * NB_ * D_) return;
  const float v = pm[idx] + gate[0] * dctx[idx];
  pm[idx] = v; pmh[idx] = (_Float16)v;
}

__global__ void bcast_add_kernel(const float* __restrict__ x, const float* __restrict__ y,
                                 float* __restrict__ out) {
  const int idx = blockIdx.x * 256 + threadIdx.x;
  if (idx >= BB * NN * D_) return;
  const int b = idx / (NN * D_);
  const int d = idx % D_;
  out[idx] = x[idx] + y[b * D_ + d];
}

__global__ void f32_to_f16_kernel(const float* __restrict__ x, _Float16* __restrict__ y, int n) {
  const int i = blockIdx.x * 256 + threadIdx.x;
  if (i < n) y[i] = (_Float16)x[i];
}

static inline int cdiv_i(int a, int b) { return (a + b - 1) / b; }

extern "C" void kernel_launch(void* const* d_in, const int* in_sizes, int n_in,
                              void* d_out, int out_size, void* d_ws, size_t ws_size,
                              hipStream_t stream) {
  (void)in_sizes; (void)n_in; (void)out_size; (void)ws_size;
  // setup_inputs() dict, recursive insertion-order flattening:
  const float* x        = (const float*)d_in[0];
  const float* pe_ww    = (const float*)d_in[1];
  const float* pe_wb    = (const float*)d_in[2];
  const float* pe_projw = (const float*)d_in[3];
  const float* pe_projb = (const float*)d_in[4];
  const float* pe_ng    = (const float*)d_in[5];
  const float* pe_nbta  = (const float*)d_in[6];
  const float* pe_pos   = (const float*)d_in[7];
  const float* qp_w     = (const float*)d_in[8];
  const float* qp_b     = (const float*)d_in[9];
  const float* kp_w     = (const float*)d_in[10];
  const float* kp_b     = (const float*)d_in[11];
  const float* fo_inw   = (const float*)d_in[12];
  const float* fo_inb   = (const float*)d_in[13];
  const float* fo_outw  = (const float*)d_in[14];
  const float* fo_outb  = (const float*)d_in[15];
  const float* n1_g     = (const float*)d_in[16];
  const float* n1_b     = (const float*)d_in[17];
  const float* n2_g     = (const float*)d_in[18];
  const float* n2_b     = (const float*)d_in[19];
  const float* ff1_w    = (const float*)d_in[20];
  const float* ff1_b    = (const float*)d_in[21];
  const float* ff2_w    = (const float*)d_in[22];
  const float* ff2_b    = (const float*)d_in[23];
  const float* ma_inw   = (const float*)d_in[24];
  const float* ma_inb   = (const float*)d_in[25];
  const float* ma_outw  = (const float*)d_in[26];
  const float* ma_outb  = (const float*)d_in[27];
  const float* mn_g     = (const float*)d_in[28];
  const float* mn_b     = (const float*)d_in[29];
  const float* mg1_w    = (const float*)d_in[30];
  const float* mg1_b    = (const float*)d_in[31];
  const float* mg2_w    = (const float*)d_in[32];
  const float* mg2_b    = (const float*)d_in[33];
  const float* op_w     = (const float*)d_in[34];
  const float* op_b     = (const float*)d_in[35];
  const float* on_g     = (const float*)d_in[36];
  const float* on_b     = (const float*)d_in[37];
  const float* ln1_g    = (const float*)d_in[38];
  const float* ln1_b    = (const float*)d_in[39];
  const float* ln2_g    = (const float*)d_in[40];
  const float* ln2_b    = (const float*)d_in[41];
  const float* mlp1_w   = (const float*)d_in[42];
  const float* mlp1_b   = (const float*)d_in[43];
  const float* mlp2_w   = (const float*)d_in[44];
  const float* mlp2_b   = (const float*)d_in[45];
  float* out = (float*)d_out;

  // Workspace bump allocator (256B aligned).
  char* wsb = (char*)d_ws;
  size_t off = 0;
  auto alloc = [&](size_t bytes) -> char* {
    char* p = wsb + off;
    off = (off + bytes + 255) & ~(size_t)255;
    return p;
  };
  float*    hf      = (float*)   alloc((size_t)BB * NN * D_ * 4);
  _Float16* hh      = (_Float16*)alloc((size_t)BB * NN * D_ * 2);
  _Float16* featin  = (_Float16*)alloc((size_t)BB * NB_ * 3 * D_ * 2);
  float*    featbuf = (float*)   alloc((size_t)BB * NB_ * D_ * 4);
  float*    pmf     = (float*)   alloc((size_t)BB * NB_ * D_ * 4);
  _Float16* pmh     = (_Float16*)alloc((size_t)BB * NB_ * D_ * 2);
  _Float16* pmnh    = (_Float16*)alloc((size_t)BB * NB_ * D_ * 2);
  float*    stf     = (float*)   alloc((size_t)BB * D_ * 4);
  float*    qsf     = (float*)   alloc((size_t)BB * D_ * 4);
  float*    kpall   = (float*)   alloc((size_t)BB * NB_ * D_ * 4);
  int*      starts  = (int*)     alloc(256);
  float*    ctxf    = (float*)   alloc((size_t)BB * CTXMAX * D_ * 4);
  _Float16* ctxh    = (_Float16*)alloc((size_t)BB * CTXMAX * D_ * 2);
  _Float16* nctxh   = (_Float16*)alloc((size_t)BB * CTXMAX * D_ * 2);
  float*    qkvbuf  = (float*)   alloc((size_t)BB * CTXMAX * 3 * D_ * 4);
  float*    o0f     = (float*)   alloc((size_t)BB * D_ * 4);
  float*    clf     = (float*)   alloc((size_t)BB * D_ * 4);
  float*    clnf    = (float*)   alloc((size_t)BB * D_ * 4);
  float*    ffhid   = (float*)   alloc((size_t)BB * FF_ * 4);
  float*    qmaf    = (float*)   alloc((size_t)BB * NB_ * D_ * 4);
  float*    kmaf    = (float*)   alloc((size_t)BB * CTXMAX * D_ * 4);
  float*    vmaf    = (float*)   alloc((size_t)BB * CTXMAX * D_ * 4);
  _Float16* aoh     = (_Float16*)alloc((size_t)BB * NB_ * D_ * 2);
  float*    dctxf   = (float*)   alloc((size_t)BB * NB_ * D_ * 4);
  float*    gatef   = (float*)   alloc(256);
  float*    stnf    = (float*)   alloc((size_t)BB * D_ * 4);
  float*    ybuf    = (float*)   alloc((size_t)BB * D_ * 4);
  _Float16* xnh     = (_Float16*)alloc((size_t)BB * NN * D_ * 2);
  _Float16* hiddenh = (_Float16*)alloc((size_t)BB * NN * FF_ * 2);
  // f16 weight copies (only matrices that feed WMMA GEMMs)
  _Float16* projh  = (_Float16*)alloc((size_t)D_ * 3 * D_ * 2);
  _Float16* kph    = (_Float16*)alloc((size_t)D_ * D_ * 2);
  _Float16* foinh  = (_Float16*)alloc((size_t)3 * D_ * D_ * 2);
  _Float16* mainh  = (_Float16*)alloc((size_t)3 * D_ * D_ * 2);
  _Float16* maouth = (_Float16*)alloc((size_t)D_ * D_ * 2);
  _Float16* mlp1h  = (_Float16*)alloc((size_t)FF_ * D_ * 2);
  _Float16* mlp2h  = (_Float16*)alloc((size_t)D_ * FF_ * 2);

  auto conv = [&](const float* src, _Float16* dst, int n) {
    f32_to_f16_kernel<<<cdiv_i(n, 256), 256, 0, stream>>>(src, dst, n);
  };
  auto gemm = [&](const _Float16* A, int lda, const _Float16* W, int ldw,
                  const float* bias, const float* resid, long ldres,
                  float* Cf, _Float16* Ch, int ldc, int M, int N, int K, int act) {
    dim3 g(N / 64, M / 128);
    gemm_wmma_kernel<<<g, 256, 0, stream>>>(A, lda, W, ldw, bias, resid, ldres,
                                            Cf, Ch, ldc, K, act);
  };

  // 0) weight conversions
  conv(pe_projw, projh, D_ * 3 * D_);
  conv(kp_w, kph, D_ * D_);
  conv(fo_inw, foinh, 3 * D_ * D_);
  conv(ma_inw, mainh, 3 * D_ * D_);
  conv(ma_outw, maouth, D_ * D_);
  conv(mlp1_w, mlp1h, FF_ * D_);
  conv(mlp2_w, mlp2h, D_ * FF_);

  // 1) h = LN(x)
  ln512_kernel<<<BB * NN, 128, 0, stream>>>(x, ln1_g, ln1_b, hf, hh, nullptr, 1);
  // 2) peripheral stats -> [mn|sd|mx] f16
  peripheral_kernel<<<BB * NB_, 64, 0, stream>>>(hf, pe_ww, pe_wb, featin);
  // 3) feat = featin @ proj_w^T + proj_b   (WMMA, M=4096,N=512,K=1536)
  gemm(featin, 3 * D_, projh, 3 * D_, pe_projb, nullptr, 0,
       featbuf, nullptr, D_, BB * NB_, D_, 3 * D_, 0);
  // 4) pm = LN(feat) + pos[j]
  ln512_kernel<<<BB * NB_, 128, 0, stream>>>(featbuf, pe_ng, pe_nbta, pmf, pmh, pe_pos, NB_);
  // 5) st = mean over blocks
  st_mean_kernel<<<cdiv_i(BB * D_, 256), 256, 0, stream>>>(pmf, stf);

  (void)hipMemsetAsync(ctxf, 0, (size_t)BB * CTXMAX * D_ * 4, stream);
  (void)hipMemsetAsync(ctxh, 0, (size_t)BB * CTXMAX * D_ * 2, stream);

  for (int t = 0; t < NS_; ++t) {
    const int Lctx = 1 + (t + 1) * WS_;
    const int Lk = (t + 1) * WS_;
    // controller
    set_ctx_st_kernel<<<cdiv_i(BB * D_, 256), 256, 0, stream>>>(stf, ctxf, ctxh);
    small_linear_kernel<<<cdiv_i(BB * D_, 256), 256, 0, stream>>>(
        stf, qp_w, qp_b, nullptr, 0, qsf, BB, D_, D_, 0);
    gemm(pmh, D_, kph, D_, kp_b, nullptr, 0, kpall, nullptr, D_, BB * NB_, D_, D_, 0);
    saccade_select_kernel<<<BB, 256, 0, stream>>>(qsf, kpall, starts);
    gather_win_kernel<<<cdiv_i(BB * WS_ * D_, 256), 256, 0, stream>>>(hf, starts, ctxf, ctxh, t);
    // fov transformer (only ctx[:,0] survives; full WMMA QKV, q0-only attention)
    ln512_kernel<<<BB * CTXMAX, 128, 0, stream>>>(ctxf, n1_g, n1_b, nullptr, nctxh, nullptr, 1);
    gemm(nctxh, D_, foinh, D_, fo_inb, nullptr, 0,
         qkvbuf, nullptr, 3 * D_, BB * CTXMAX, 3 * D_, D_, 0);
    fov_attn_q0_kernel<<<BB * NH_, 32, 0, stream>>>(qkvbuf, Lctx, o0f);
    small_linear_kernel<<<cdiv_i(BB * D_, 256), 256, 0, stream>>>(
        o0f, fo_outw, fo_outb, ctxf, (long)CTXMAX * D_, clf, BB, D_, D_, 0);
    ln512_kernel<<<BB, 128, 0, stream>>>(clf, n2_g, n2_b, clnf, nullptr, nullptr, 1);
    small_linear_kernel<<<cdiv_i(BB * FF_, 256), 256, 0, stream>>>(
        clnf, ff1_w, ff1_b, nullptr, 0, ffhid, BB, FF_, D_, 1);
    small_linear_kernel<<<cdiv_i(BB * D_, 256), 256, 0, stream>>>(
        ffhid, ff2_w, ff2_b, clf, (long)D_, stf, BB, D_, FF_, 0);
    // memory cross-attention pm <- attn(LN(pm), ac, ac)
    ln512_kernel<<<BB * NB_, 128, 0, stream>>>(pmf, mn_g, mn_b, nullptr, pmnh, nullptr, 1);
    gemm(pmnh, D_, mainh, D_, ma_inb, nullptr, 0, qmaf, nullptr, D_, BB * NB_, D_, D_, 0);
    gemm(ctxh, D_, mainh + (size_t)D_ * D_, D_, ma_inb + D_, nullptr, 0,
         kmaf, nullptr, D_, BB * CTXMAX, D_, D_, 0);
    gemm(ctxh, D_, mainh + (size_t)2 * D_ * D_, D_, ma_inb + 2 * D_, nullptr, 0,
         vmaf, nullptr, D_, BB * CTXMAX, D_, D_, 0);
    ma_attn_kernel<<<BB * NB_ * NH_ / 8, 256, 0, stream>>>(qmaf, kmaf, vmaf, aoh, Lk);
    gemm(aoh, D_, maouth, D_, ma_outb, nullptr, 0, dctxf, nullptr, D_, BB * NB_, D_, D_, 0);
    gate_kernel<<<1, 32, 0, stream>>>(mg1_w, mg1_b, mg2_w, mg2_b,
                                      (float)t / (float)NS_, gatef);
    pm_update_kernel<<<cdiv_i(BB * NB_ * D_, 256), 256, 0, stream>>>(pmf, pmh, dctxf, gatef);
  }

  // output projection of broadcast state + residual
  ln512_kernel<<<BB, 128, 0, stream>>>(stf, on_g, on_b, stnf, nullptr, nullptr, 1);
  small_linear_kernel<<<cdiv_i(BB * D_, 256), 256, 0, stream>>>(
      stnf, op_w, op_b, nullptr, 0, ybuf, BB, D_, D_, 0);
  bcast_add_kernel<<<cdiv_i(BB * NN * D_, 256), 256, 0, stream>>>(x, ybuf, out);
  // final MLP (the big WMMA work: 2x ~69 GFLOP)
  ln512_kernel<<<BB * NN, 128, 0, stream>>>(out, ln2_g, ln2_b, nullptr, xnh, nullptr, 1);
  gemm(xnh, D_, mlp1h, D_, mlp1_b, nullptr, 0, nullptr, hiddenh, FF_, BB * NN, FF_, D_, 1);
  gemm(hiddenh, FF_, mlp2h, FF_, mlp2_b, out, (long)D_, out, nullptr, D_, BB * NN, D_, FF_, 0);
}
